// AutoSlicingModel_15418932593325
// MI455X (gfx1250) — matrix operand order repaired and checked by reading, not compiled, not run on hardware
//
#include <hip/hip_runtime.h>
#include <hip/hip_bf16.h>
#include <math.h>

// ---------------------------------------------------------------------------
// Problem constants (from reference): B=16, S=4096, H=768, NS=256, V=16
// ---------------------------------------------------------------------------
#define BB   16
#define SS   4096
#define HH   768
#define NSS  256
#define VV   16
#define MM   (BB * NSS)          // 4096 rows through the MLPs
#define K1   (2 * HH)            // 1536
#define K2   HH                  // 768

#define KCHUNK   64              // K elements staged per LDS chunk
#define BROW     72              // padded LDS row stride (halves): bank-friendly

typedef _Float16 half8  __attribute__((ext_vector_type(8)));
typedef _Float16 half16 __attribute__((ext_vector_type(16)));
typedef float    float8 __attribute__((ext_vector_type(8)));

// ---- CDNA5 async global->LDS copy (ASYNCcnt path), via inline asm ----------
__device__ __forceinline__ void async_b128_to_lds(const void* gptr, void* lptr) {
    unsigned lds = (unsigned)(unsigned long long)(uintptr_t)lptr; // low 32b = LDS offset
    asm volatile("global_load_async_to_lds_b128 %0, %1, off"
                 :: "v"(lds), "v"(gptr) : "memory");
}
#define WAIT_ASYNCCNT_LE4() asm volatile("s_wait_asynccnt 0x4" ::: "memory")

// ---------------------------------------------------------------------------
// 1) var_emb[b,h] = mean over V gathered tokens   (f32 -> f16)
// ---------------------------------------------------------------------------
__global__ __launch_bounds__(256) void varemb_kernel(
    const float* __restrict__ hidden, const int* __restrict__ vids,
    _Float16* __restrict__ varemb)
{
    int b = blockIdx.x;
    for (int h = threadIdx.x; h < HH; h += blockDim.x) {
        float sum = 0.f;
        #pragma unroll
        for (int v = 0; v < VV; ++v) {
            int tok = vids[b * VV + v];
            sum += hidden[((size_t)b * SS + tok) * HH + h];
        }
        varemb[b * HH + h] = (_Float16)(sum * (1.0f / VV));
    }
}

// ---------------------------------------------------------------------------
// 2) feats[b*NS+s, 0:H]   = segment mean of hidden (ids sorted/contiguous)
//    feats[b*NS+s, H:2H]  = var_emb[b]             (f16, row-major [M, 2H])
// ---------------------------------------------------------------------------
__global__ __launch_bounds__(256) void feats_kernel(
    const float* __restrict__ hidden, const int* __restrict__ sid,
    const _Float16* __restrict__ varemb, _Float16* __restrict__ feats)
{
    int b = blockIdx.x / NSS;
    int s = blockIdx.x % NSS;
    const int* ids = sid + (size_t)b * SS;

    int lo = 0, hi = SS;
    while (lo < hi) { int m = (lo + hi) >> 1; if (ids[m] <  s) lo = m + 1; else hi = m; }
    int lo2 = lo, hi2 = SS;
    while (lo2 < hi2) { int m = (lo2 + hi2) >> 1; if (ids[m] <= s) lo2 = m + 1; else hi2 = m; }

    int   cnt = lo2 - lo;
    float inv = 1.0f / (float)(cnt > 0 ? cnt : 1);

    const float* hb = hidden + (size_t)b * SS * HH;
    _Float16* fr = feats + (size_t)blockIdx.x * K1;
    for (int h = threadIdx.x; h < HH; h += blockDim.x) {
        float sum = 0.f;
        for (int t = lo; t < lo2; ++t) sum += hb[(size_t)t * HH + h];
        fr[h]      = (_Float16)(sum * inv);
        fr[HH + h] = varemb[b * HH + h];
    }
}

// ---------------------------------------------------------------------------
// 3) Convert + transpose weights: f32 [K,N] -> f16 [N,K]
// ---------------------------------------------------------------------------
__global__ __launch_bounds__(256) void convT_kernel(
    const float* __restrict__ W, _Float16* __restrict__ Wt, int K, int N)
{
    int idx = blockIdx.x * blockDim.x + threadIdx.x;
    if (idx >= K * N) return;
    int n = idx / K, k = idx % K;
    Wt[(size_t)n * K + k] = (_Float16)W[(size_t)k * N + n];
}

// ---------------------------------------------------------------------------
// 4) WMMA GEMM + bias + exact GELU:  Y[M,N] = gelu(A[M,K] @ Wt[N,K]^T + bias)
//    block = 128 threads (4 waves); wave tile = 16(M) x 64(N).
//    B tile (64 cols x KCHUNK) staged in LDS with async copies, double-
//    buffered so chunk i+1 streams in (ASYNCcnt) while chunk i feeds WMMA.
//    All 16 ds_load_b128 of a chunk are pinned BEFORE the 8-WMMA chain via
//    sched_barrier so DS latency overlaps the matrix pipe (in-order DS
//    returns => incremental s_wait_dscnt instead of 0 per WMMA).
// ---------------------------------------------------------------------------
__global__ __launch_bounds__(128) void gemm_wmma_kernel(
    const _Float16* __restrict__ A,   // [M, K]
    const _Float16* __restrict__ Wt,  // [N, K]
    const float*    __restrict__ bias,// [N]
    _Float16*       __restrict__ Y,   // [M, N]
    int M, int N, int K)
{
    __shared__ _Float16 shB[2][64 * BROW];   // ~18 KB, double-buffered B tile

    const int tid   = threadIdx.x;
    const int lane  = tid & 31;
    const int wave  = tid >> 5;
    const int l15   = lane & 15;
    const int hi    = lane >> 4;             // 0 or 1
    const int tileM = blockIdx.y * 64 + wave * 16;
    const int tileN = blockIdx.x * 64;

    const _Float16* aRow = A + (size_t)(tileM + l15) * K;

    // B staging: thread t covers row r = t&63, half-part p = t>>6 (64B = 4 x b128)
    const int r = tid & 63;
    const int p = tid >> 6;
    const _Float16* gB = Wt + (size_t)(tileN + r) * K + p * 32;
    _Float16* lB0 = &shB[0][r * BROW + p * 32];
    _Float16* lB1 = &shB[1][r * BROW + p * 32];

    const int nCh = K / KCHUNK;

    // prologue: chunk 0 -> buffer 0
    #pragma unroll
    for (int q = 0; q < 4; ++q)
        async_b128_to_lds(gB + q * 8, lB0 + q * 8);

    float8 acc[4] = {};

    for (int i = 0; i < nCh; ++i) {
        // all waves are done reading buf[(i+1)&1] (consumed in iteration i-1)
        __syncthreads();

        // stream chunk i+1 into the alternate buffer (wrap on last iter: dead
        // writes into the non-consumed buffer, drained by s_endpgm)
        {
            const _Float16* g = gB + (size_t)((i + 1 == nCh) ? 0 : i + 1) * KCHUNK;
            _Float16* l = ((i + 1) & 1) ? lB1 : lB0;
            #pragma unroll
            for (int q = 0; q < 4; ++q)
                async_b128_to_lds(g + q * 8, l + q * 8);
        }

        // A fragments for both K-steps of this chunk (global, overlaps wait)
        const int kbase = i * KCHUNK;
        half8 a00 = *(const half8*)(aRow + kbase      + hi * 16);
        half8 a01 = *(const half8*)(aRow + kbase +  8 + hi * 16);
        half8 a10 = *(const half8*)(aRow + kbase + 32 + hi * 16);
        half8 a11 = *(const half8*)(aRow + kbase + 40 + hi * 16);
        half16 aF0 = __builtin_shufflevector(a00, a01,
                       0,1,2,3,4,5,6,7,8,9,10,11,12,13,14,15);
        half16 aF1 = __builtin_shufflevector(a10, a11,
                       0,1,2,3,4,5,6,7,8,9,10,11,12,13,14,15);
        if (i + 1 < nCh) __builtin_prefetch(aRow + kbase + KCHUNK, 0, 3);

        // chunk i has landed in LDS for this wave; barrier makes it WG-visible
        WAIT_ASYNCCNT_LE4();
        __syncthreads();

        // ---- preload ALL 8 B fragments (16 ds_load_b128 back-to-back) ----
        const _Float16* bbuf = (i & 1) ? &shB[1][0] : &shB[0][0];
        const _Float16* bl0  = bbuf + l15 * BROW + hi * 16;
        half16 bF[8];
        #pragma unroll
        for (int ks = 0; ks < 2; ++ks) {
            #pragma unroll
            for (int j = 0; j < 4; ++j) {
                const _Float16* bl = bl0 + j * (16 * BROW) + ks * 32;
                half8 b0 = *(const half8*)(bl);
                half8 b1 = *(const half8*)(bl + 8);
                bF[ks * 4 + j] = __builtin_shufflevector(b0, b1,
                                   0,1,2,3,4,5,6,7,8,9,10,11,12,13,14,15);
            }
        }
        // pin: no ds_load may sink past this point, no WMMA may hoist above
        __builtin_amdgcn_sched_barrier(0);

        // ---- dense WMMA chain: 8 ops, 4 independent accumulators ----
        #pragma unroll
        for (int ks = 0; ks < 2; ++ks) {
            half16 aF = ks ? aF1 : aF0;
            #pragma unroll
            for (int j = 0; j < 4; ++j) {
                acc[j] = __builtin_amdgcn_wmma_f32_16x16x32_f16(
                            false, aF, false, bF[ks * 4 + j],
                            (short)0, acc[j], false, false);
            }
        }
        __builtin_amdgcn_sched_barrier(0);
    }

    // epilogue: bias + exact GELU, store f16
    #pragma unroll
    for (int j = 0; j < 4; ++j) {
        int col = tileN + j * 16 + l15;
        float bv = bias[col];
        #pragma unroll
        for (int rr = 0; rr < 8; ++rr) {
            int row = tileM + rr + hi * 8;
            float x = acc[j][rr] + bv;
            float g = 0.5f * x * (1.0f + erff(x * 0.70710678118654752f));
            Y[(size_t)row * N + col] = (_Float16)g;
        }
    }
}

// ---------------------------------------------------------------------------
// 5) Final head: out[head, b, s] = mask * (dot(h2[row], w3) + b3)
//    one wave32 per (head,row); shuffle reduction.
// ---------------------------------------------------------------------------
__global__ __launch_bounds__(256) void head_kernel(
    const _Float16* __restrict__ H2b, const _Float16* __restrict__ H2f,
    const float* __restrict__ w3b, const float* __restrict__ b3b,
    const float* __restrict__ w3f, const float* __restrict__ b3f,
    const int* __restrict__ line_nums, float* __restrict__ out)
{
    int gwave = (blockIdx.x * blockDim.x + threadIdx.x) >> 5;
    int lane  = threadIdx.x & 31;
    if (gwave >= 2 * MM) return;
    int head = gwave / MM;
    int row  = gwave % MM;
    int b = row / NSS, s = row % NSS;

    const _Float16* h2 = (head ? H2f : H2b) + (size_t)row * HH;
    const float*    w3 = head ? w3f : w3b;

    float acc = 0.f;
    for (int k = lane; k < HH; k += 32) acc += (float)h2[k] * w3[k];
    #pragma unroll
    for (int off = 16; off > 0; off >>= 1) acc += __shfl_down(acc, off, 32);

    if (lane == 0) {
        float val = acc + (head ? b3f[0] : b3b[0]);
        int   ln  = line_nums[b];
        bool  keep = head ? (s > ln) : (s < ln);
        out[(size_t)head * MM + row] = keep ? val : 0.0f;
    }
}

// ---------------------------------------------------------------------------
// Launch
// ---------------------------------------------------------------------------
extern "C" void kernel_launch(void* const* d_in, const int* in_sizes, int n_in,
                              void* d_out, int out_size, void* d_ws, size_t ws_size,
                              hipStream_t stream) {
    const float* hidden = (const float*)d_in[0];
    const float* b_w1 = (const float*)d_in[1];  const float* b_b1 = (const float*)d_in[2];
    const float* b_w2 = (const float*)d_in[3];  const float* b_b2 = (const float*)d_in[4];
    const float* b_w3 = (const float*)d_in[5];  const float* b_b3 = (const float*)d_in[6];
    const float* f_w1 = (const float*)d_in[7];  const float* f_b1 = (const float*)d_in[8];
    const float* f_w2 = (const float*)d_in[9];  const float* f_b2 = (const float*)d_in[10];
    const float* f_w3 = (const float*)d_in[11]; const float* f_b3 = (const float*)d_in[12];
    const int* statements_ids = (const int*)d_in[13];
    const int* variables_ids  = (const int*)d_in[14];
    const int* line_nums      = (const int*)d_in[15];
    float* out = (float*)d_out;

    // ---- workspace layout (bytes, 256-aligned) ----
    char* ws = (char*)d_ws;
    size_t off = 0;
    auto alloc = [&](size_t bytes) { size_t o = off; off += (bytes + 255) & ~(size_t)255; return o; };

    size_t off_var   = alloc((size_t)BB * HH * 2);          // f16 [B,H]
    size_t off_feats = alloc((size_t)MM * K1 * 2);          // f16 [M,2H]
    size_t off_w1t_b = alloc((size_t)HH * K1 * 2);          // f16 [N=768,K=1536]
    size_t off_w1t_f = alloc((size_t)HH * K1 * 2);
    size_t off_w2t_b = alloc((size_t)HH * K2 * 2);          // f16 [768,768]
    size_t off_w2t_f = alloc((size_t)HH * K2 * 2);
    size_t off_h1_b  = alloc((size_t)MM * HH * 2);          // f16 [M,768]
    size_t off_h1_f  = alloc((size_t)MM * HH * 2);
    // h2 reuses the feats region (feats dead after layer-1 GEMMs)
    size_t off_h2_b  = off_feats;
    size_t off_h2_f  = off_feats + (size_t)MM * HH * 2;

    _Float16* varemb = (_Float16*)(ws + off_var);
    _Float16* feats  = (_Float16*)(ws + off_feats);
    _Float16* w1t_b  = (_Float16*)(ws + off_w1t_b);
    _Float16* w1t_f  = (_Float16*)(ws + off_w1t_f);
    _Float16* w2t_b  = (_Float16*)(ws + off_w2t_b);
    _Float16* w2t_f  = (_Float16*)(ws + off_w2t_f);
    _Float16* h1_b   = (_Float16*)(ws + off_h1_b);
    _Float16* h1_f   = (_Float16*)(ws + off_h1_f);
    _Float16* h2_b   = (_Float16*)(ws + off_h2_b);
    _Float16* h2_f   = (_Float16*)(ws + off_h2_f);

    // 1) variable embedding
    varemb_kernel<<<BB, 256, 0, stream>>>(hidden, variables_ids, varemb);

    // 2) features (segment means + broadcast var_emb), f16
    feats_kernel<<<BB * NSS, 256, 0, stream>>>(hidden, statements_ids, varemb, feats);

    // 3) weight convert + transpose
    {
        int n1 = K1 * HH, n2 = K2 * HH;
        convT_kernel<<<(n1 + 255) / 256, 256, 0, stream>>>(b_w1, w1t_b, K1, HH);
        convT_kernel<<<(n1 + 255) / 256, 256, 0, stream>>>(f_w1, w1t_f, K1, HH);
        convT_kernel<<<(n2 + 255) / 256, 256, 0, stream>>>(b_w2, w2t_b, K2, HH);
        convT_kernel<<<(n2 + 255) / 256, 256, 0, stream>>>(f_w2, w2t_f, K2, HH);
    }

    // 4) layer 1: [4096,1536] @ [1536,768] -> gelu -> h1
    {
        dim3 grid(HH / 64, MM / 64);   // (12, 64)
        gemm_wmma_kernel<<<grid, 128, 0, stream>>>(feats, w1t_b, b_b1, h1_b, MM, HH, K1);
        gemm_wmma_kernel<<<grid, 128, 0, stream>>>(feats, w1t_f, f_b1, h1_f, MM, HH, K1);
    }

    // 5) layer 2: [4096,768] @ [768,768] -> gelu -> h2 (reuses feats region)
    {
        dim3 grid(HH / 64, MM / 64);
        gemm_wmma_kernel<<<grid, 128, 0, stream>>>(h1_b, w2t_b, b_b2, h2_b, MM, HH, K2);
        gemm_wmma_kernel<<<grid, 128, 0, stream>>>(h1_f, w2t_f, f_b2, h2_f, MM, HH, K2);
    }

    // 6) final projection + masking -> out [2, B, NS]
    {
        int waves = 2 * MM;                       // 8192 rows
        int blocks = (waves * 32 + 255) / 256;    // 8 waves/block
        head_kernel<<<blocks, 256, 0, stream>>>(h2_b, h2_f, b_w3, b_b3, f_w3, f_b3,
                                                line_nums, out);
    }
}
